// PhiMambaDecoderLayer_34437047779932
// MI455X (gfx1250) — matrix operand order repaired and checked by reading, not compiled, not run on hardware
//
#include <hip/hip_runtime.h>
#include <hip/hip_bf16.h>
#include <stdint.h>

#define B_SZ   2
#define SEQ    2048
#define DM     2048
#define DI     4096
#define DXB    1024
#define INTER  8192
#define HD     64
#define DSN    64
#define NH     64
#define DP     10304
#define ROWS   (B_SZ*SEQ)      /* 4096 */
#define EPSF   1e-5f
#define LDT    40              /* padded LDS row stride (bf16 elems) */

typedef __attribute__((ext_vector_type(16))) __bf16        v16bf;
typedef __attribute__((ext_vector_type(8)))  float         v8f;
typedef __attribute__((ext_vector_type(4)))  unsigned int  v4u;
typedef __attribute__((ext_vector_type(4)))  unsigned int  u32x4;
typedef __attribute__((ext_vector_type(8)))  int           i32x8;
typedef __attribute__((ext_vector_type(4)))  int           i32x4;

#if __has_builtin(__builtin_amdgcn_tensor_load_to_lds) && __has_builtin(__builtin_amdgcn_s_wait_tensorcnt)
#define USE_TDM 1
#else
#define USE_TDM 0
#endif

union Frag { v4u q[2]; v16bf v; };

__device__ __forceinline__ unsigned short f2bf(float f) {
  unsigned int u = __float_as_uint(f);
  u += 0x7FFFu + ((u >> 16) & 1u);          // round-to-nearest-even
  return (unsigned short)(u >> 16);
}

// ---------------------------------------------------------------------------
// LayerNorm: one block per row of 2048; writes bf16 activations for the GEMMs
// ---------------------------------------------------------------------------
__global__ __launch_bounds__(256)
void ln_kernel(const float* __restrict__ hs, const float* __restrict__ w,
               const float* __restrict__ b, unsigned short* __restrict__ outb)
{
  __shared__ float red[256];
  const int row = blockIdx.x;
  const int t = threadIdx.x;
  const float* x = hs + (size_t)row * DM;
  float v[8]; float s = 0.f;
  #pragma unroll
  for (int i = 0; i < 8; ++i) { v[i] = x[t + i * 256]; s += v[i]; }
  red[t] = s; __syncthreads();
  for (int k = 128; k > 0; k >>= 1) { if (t < k) red[t] += red[t + k]; __syncthreads(); }
  const float mu = red[0] * (1.f / DM);
  __syncthreads();
  float q = 0.f;
  #pragma unroll
  for (int i = 0; i < 8; ++i) { float d = v[i] - mu; q += d * d; }
  red[t] = q; __syncthreads();
  for (int k = 128; k > 0; k >>= 1) { if (t < k) red[t] += red[t + k]; __syncthreads(); }
  const float rstd = rsqrtf(red[0] * (1.f / DM) + EPSF);
  #pragma unroll
  for (int i = 0; i < 8; ++i) {
    const int j = t + i * 256;
    outb[(size_t)row * DM + j] = f2bf((v[i] - mu) * rstd * w[j] + b[j]);
  }
}

// ---------------------------------------------------------------------------
// bf16 WMMA GEMM:  C[M,N] f32  =  A[M,K] bf16  @  W[K,N] f32 (converted inline)
// 256 threads (8 waves), 128x128 block tile, BK=32, wave tile 32x64 (2x4 wmma).
// A tile staged by the Tensor Data Mover (tensor_load_to_lds, padded to LDT);
// W tile converted f32->bf16 and transposed with packed b32 stores.
// ---------------------------------------------------------------------------
__global__ __launch_bounds__(256)
void gemm_bf16f32(const unsigned short* __restrict__ A,
                  const float* __restrict__ W,
                  float* __restrict__ C,
                  int M, int N, int K)
{
  // one static shared block: sA at LDS offset 0 (TDM descriptor relies on it)
  __shared__ __align__(16) unsigned short smem[128 * LDT * 2];
  unsigned short* sA = smem;
  unsigned short* sB = smem + 128 * LDT;

  const int t    = threadIdx.x;
  const int nBlk = blockIdx.x, mBlk = blockIdx.y;
  const int lane = t & 31, wave = t >> 5;
  const int mBase = (wave >> 1) * 32;     // 4 wave-rows
  const int nBase = (wave & 1) * 64;      // 2 wave-cols
  const int lr = lane & 15, kh = lane >> 4;

  v8f acc[2][4] = {};                     // zero-init accumulators

  // W-stage mapping: each thread packs k-pair (kp,kp+1) x 8 n-values
  const int kp = (t >> 4) * 2;            // even k: 0..30
  const int nb = (t & 15) * 8;            // n offset: 0..120

  for (int ko = 0; ko < K; ko += 32) {
#if USE_TDM
    // --- A tile via Tensor Data Mover: 2D tile 32(elems) x 128(rows),
    //     element 2B, tensor stride K, LDS pad 4 DW every 16 DW -> stride LDT
    if (t < 32) {
      unsigned long long ga = (unsigned long long)(uintptr_t)A
          + ((((unsigned long long)(mBlk * 128)) * (unsigned long long)(unsigned)K
              + (unsigned long long)(unsigned)ko) << 1);
      u32x4 g0 = { 1u,                                  // count=1, user desc
                   0u,                                  // lds_addr = sA @ 0
                   (unsigned)ga,
                   (unsigned)((ga >> 32) & 0x01FFFFFFull) | 0x80000000u }; // type=2
      i32x8 g1 = { (int)0x06D10000,                     // data_size=2B, pad_en,
                                                        // pad_interval=16DW, pad=4DW
                   (int)(((unsigned)K & 0xFFFFu) << 16),            // tensor_dim0 lo
                   (int)((((unsigned)M & 0xFFFFu) << 16) | ((unsigned)K >> 16)),
                   (int)((32u << 16) | ((unsigned)M >> 16)),        // tile_dim0=32
                   128,                                  // tile_dim1=128
                   K,                                    // tensor_dim0_stride lo
                   0, 0 };
      i32x4 gz4 = { 0, 0, 0, 0 };
      i32x8 gz8 = { 0, 0, 0, 0, 0, 0, 0, 0 };
      __builtin_amdgcn_tensor_load_to_lds(g0, g1, gz4, gz4, gz8, 0);
    }
#else
    {
      const int ar = t >> 1;
      const int ac = (t & 1) * 16;
      const v4u* pa = (const v4u*)(A + (size_t)(mBlk * 128 + ar) * K + ko + ac);
      *(v4u*)&sA[ar * LDT + ac]     = pa[0];
      *(v4u*)&sA[ar * LDT + ac + 8] = pa[1];
    }
#endif
    // --- stage W tile: f32 -> bf16, transposed into sB (row = n, col = k),
    //     two k's packed per b32 store
    {
      const float* w0r = W + (size_t)(ko + kp) * N;
      const float* w1r = w0r + N;
      const int gn0 = nBlk * 128 + nb;
      float a0[8], a1[8];
      if (gn0 + 8 <= N) {
        const float4* p0 = (const float4*)(w0r + gn0);
        const float4* p1 = (const float4*)(w1r + gn0);
        float4 x0 = p0[0], x1 = p0[1], y0 = p1[0], y1 = p1[1];
        a0[0]=x0.x; a0[1]=x0.y; a0[2]=x0.z; a0[3]=x0.w;
        a0[4]=x1.x; a0[5]=x1.y; a0[6]=x1.z; a0[7]=x1.w;
        a1[0]=y0.x; a1[1]=y0.y; a1[2]=y0.z; a1[3]=y0.w;
        a1[4]=y1.x; a1[5]=y1.y; a1[6]=y1.z; a1[7]=y1.w;
      } else {
        #pragma unroll
        for (int i = 0; i < 8; ++i) {
          a0[i] = (gn0 + i < N) ? w0r[gn0 + i] : 0.f;
          a1[i] = (gn0 + i < N) ? w1r[gn0 + i] : 0.f;
        }
      }
      #pragma unroll
      for (int i = 0; i < 8; ++i) {
        const unsigned pk = (unsigned)f2bf(a0[i]) | ((unsigned)f2bf(a1[i]) << 16);
        *(unsigned*)&sB[(nb + i) * LDT + kp] = pk;
      }
    }
#if USE_TDM
    if (t < 32) __builtin_amdgcn_s_wait_tensorcnt(0);
#endif
    __syncthreads();

    // fragment loads per ISA 16-bit A/B layouts (lanes 0-15: K-low, 16-31: K-high)
    Frag aF[2], bF[4];
    #pragma unroll
    for (int mt = 0; mt < 2; ++mt) {
      const int r = mBase + mt * 16 + lr;
      aF[mt].q[0] = *(const v4u*)&sA[r * LDT + kh * 8];
      aF[mt].q[1] = *(const v4u*)&sA[r * LDT + 16 + kh * 8];
    }
    #pragma unroll
    for (int nt = 0; nt < 4; ++nt) {
      const int r = nBase + nt * 16 + lr;
      bF[nt].q[0] = *(const v4u*)&sB[r * LDT + kh * 8];
      bF[nt].q[1] = *(const v4u*)&sB[r * LDT + 16 + kh * 8];
    }
    #pragma unroll
    for (int nt = 0; nt < 4; ++nt)
      #pragma unroll
      for (int mt = 0; mt < 2; ++mt)
        acc[mt][nt] = __builtin_amdgcn_wmma_f32_16x16x32_bf16(
            false, aF[mt].v, false, bF[nt].v, (short)0, acc[mt][nt], false, false);
    __syncthreads();
  }

  // epilogue: C/D layout -> M = vgpr + 8*(lane/16), N = lane%16
  #pragma unroll
  for (int mt = 0; mt < 2; ++mt)
    #pragma unroll
    for (int nt = 0; nt < 4; ++nt) {
      const int gc = nBlk * 128 + nBase + nt * 16 + lr;
      if (gc < N) {
        #pragma unroll
        for (int e = 0; e < 8; ++e) {
          const int gr = mBlk * 128 + mBase + mt * 16 + kh * 8 + e;
          C[(size_t)gr * N + gc] = acc[mt][nt][e];
        }
      }
    }
}

// ---------------------------------------------------------------------------
// Selective scan: one block per (batch, head); 64x64 f32 state in registers
// (16 per thread), x/B/C staged to LDS each step, quad shuffle-reduce for y.
// ---------------------------------------------------------------------------
__global__ __launch_bounds__(256)
void scan_kernel(const float* __restrict__ zx, const float* __restrict__ dt_bias,
                 const float* __restrict__ A_log, const float* __restrict__ Dp,
                 float* __restrict__ y)
{
  const int bh = blockIdx.x;
  const int b = bh >> 6;                 // / NH
  const int h = bh & 63;
  const int t = threadIdx.x;
  const int p = t >> 2;                  // head-dim index (0..63)
  const int n0 = (t & 3) * 16;           // state-dim chunk

  const float Ah = -__expf(A_log[h]);
  const float db = dt_bias[h];
  const float Dh = Dp[h];
  const int kv = h >> 2;                 // GQA kv head

  __shared__ float sx[HD], sB[DSN], sC[DSN];
  __shared__ float sdt;

  float st[16];
  #pragma unroll
  for (int i = 0; i < 16; ++i) st[i] = 0.f;

  size_t base = (size_t)(b * SEQ) * DP;
  for (int l = 0; l < SEQ; ++l, base += DP) {
    if (t < 64)        sx[t]       = zx[base + DI + kv * HD + t];
    else if (t < 128)  sB[t - 64]  = zx[base + DI + DXB + kv * DSN + (t - 64)];
    else if (t < 192)  sC[t - 128] = zx[base + DI + 2*DXB + h * DSN + (t - 128)];
    else if (t == 192) sdt         = zx[base + DI + 2*DXB + DI + h];
    __syncthreads();

    const float u   = sdt + db;
    const float dtv = (u > 20.f) ? u : log1pf(__expf(u));
    const float dA  = __expf(dtv * Ah);
    const float xv  = sx[p];
    const float dbx = dtv * xv;

    float acc = 0.f;
    #pragma unroll
    for (int i = 0; i < 16; ++i) {
      float s = st[i] * dA + dbx * sB[n0 + i];
      st[i] = s;
      acc += s * sC[n0 + i];
    }
    acc += __shfl_xor(acc, 1, 32);
    acc += __shfl_xor(acc, 2, 32);
    if ((t & 3) == 0)
      y[(size_t)(b * SEQ + l) * DI + h * HD + p] = acc + Dh * xv;
    __syncthreads();
  }
}

// ---------------------------------------------------------------------------
// Gate (silu) + RMSNorm, writes bf16 for the out-projection GEMM
// ---------------------------------------------------------------------------
__global__ __launch_bounds__(256)
void gate_kernel(const float* __restrict__ zx, const float* __restrict__ y,
                 const float* __restrict__ norm_w, unsigned short* __restrict__ yn)
{
  __shared__ float red[256];
  const int row = blockIdx.x;
  const int t = threadIdx.x;
  const size_t zb = (size_t)row * DP;
  const size_t ybo = (size_t)row * DI;
  float vg[16]; float ss = 0.f;
  #pragma unroll
  for (int i = 0; i < 16; ++i) {
    const int j = t + i * 256;
    const float z = zx[zb + j];
    const float g = y[ybo + j] * (z / (1.f + __expf(-z)));
    vg[i] = g; ss += g * g;
  }
  red[t] = ss; __syncthreads();
  for (int s = 128; s > 0; s >>= 1) { if (t < s) red[t] += red[t + s]; __syncthreads(); }
  const float rs = rsqrtf(red[0] * (1.f / DI) + EPSF);
  #pragma unroll
  for (int i = 0; i < 16; ++i) {
    const int j = t + i * 256;
    yn[ybo + j] = f2bf(vg[i] * rs * norm_w[j]);
  }
}

// ---------------------------------------------------------------------------
// Bias + tanh-GELU, fp32 -> bf16 for the fc2 GEMM
// ---------------------------------------------------------------------------
__global__ __launch_bounds__(256)
void gelu_kernel(const float* __restrict__ in, const float* __restrict__ bias,
                 unsigned short* __restrict__ outb)
{
  const size_t i4 = ((size_t)blockIdx.x * blockDim.x + threadIdx.x) * 4;
  if (i4 >= (size_t)ROWS * INTER) return;
  const float4 f = *(const float4*)(in + i4);
  float v[4] = { f.x, f.y, f.z, f.w };
  #pragma unroll
  for (int k = 0; k < 4; ++k) {
    const int col = (int)((i4 + k) & (INTER - 1));
    const float x = v[k] + bias[col];
    const float g = 0.5f * x * (1.f + tanhf(0.7978845608028654f * (x + 0.044715f * x * x * x)));
    outb[i4 + k] = f2bf(g);
  }
}

// ---------------------------------------------------------------------------
// out = attn + (mlp + fc2_b) + residual
// ---------------------------------------------------------------------------
__global__ __launch_bounds__(256)
void final_kernel(const float* __restrict__ attn, const float* __restrict__ mlp,
                  const float* __restrict__ fc2_b, const float* __restrict__ resid,
                  float* __restrict__ out)
{
  const size_t i4 = ((size_t)blockIdx.x * blockDim.x + threadIdx.x) * 4;
  if (i4 >= (size_t)ROWS * DM) return;
  const float4 a = *(const float4*)(attn + i4);
  const float4 m = *(const float4*)(mlp + i4);
  const float4 r = *(const float4*)(resid + i4);
  float4 o;
  o.x = a.x + m.x + r.x + fc2_b[(i4 + 0) & (DM - 1)];
  o.y = a.y + m.y + r.y + fc2_b[(i4 + 1) & (DM - 1)];
  o.z = a.z + m.z + r.z + fc2_b[(i4 + 2) & (DM - 1)];
  o.w = a.w + m.w + r.w + fc2_b[(i4 + 3) & (DM - 1)];
  *(float4*)(out + i4) = o;
}

// ---------------------------------------------------------------------------
extern "C" void kernel_launch(void* const* d_in, const int* in_sizes, int n_in,
                              void* d_out, int out_size, void* d_ws, size_t ws_size,
                              hipStream_t stream)
{
  (void)in_sizes; (void)n_in; (void)out_size; (void)ws_size;
  const float* hs     = (const float*)d_in[0];
  const float* ln_w   = (const float*)d_in[1];
  const float* ln_b   = (const float*)d_in[2];
  const float* W_in   = (const float*)d_in[3];
  const float* dt_b   = (const float*)d_in[4];
  const float* A_log  = (const float*)d_in[5];
  const float* Dv     = (const float*)d_in[6];
  const float* norm_w = (const float*)d_in[7];
  const float* W_out  = (const float*)d_in[8];
  const float* fc1_w  = (const float*)d_in[9];
  const float* fc1_b  = (const float*)d_in[10];
  const float* fc2_w  = (const float*)d_in[11];
  const float* fc2_b  = (const float*)d_in[12];
  float* out = (float*)d_out;

  char* ws = (char*)d_ws;
  size_t o = 0;
  unsigned short* hsln = (unsigned short*)(ws + o);                 o += (size_t)ROWS * DM * 2;
  const size_t o_zx = o;  float* zx = (float*)(ws + o);             o += (size_t)ROWS * DP * 4;
  const size_t o_y  = o;  float* yb = (float*)(ws + o);             o += (size_t)ROWS * DI * 4;
  const size_t o_yn = o;  unsigned short* yn = (unsigned short*)(ws + o); o += (size_t)ROWS * DI * 2;
  float* attn = (float*)(ws + o);
  // aliases with disjoint lifetimes:
  float*          mlph = (float*)(ws + o_zx);           // fc1 out f32 (128MB <= zx 169MB)
  unsigned short* mlpb = (unsigned short*)(ws + o_y);   // gelu out bf16 (64MB == y 64MB)
  float*          mlpo = (float*)(ws + o_yn);           // fc2 out f32 (32MB == yn 32MB)

  ln_kernel<<<ROWS, 256, 0, stream>>>(hs, ln_w, ln_b, hsln);
  gemm_bf16f32<<<dim3((DP + 127) / 128, ROWS / 128), 256, 0, stream>>>(hsln, W_in, zx, ROWS, DP, DM);
  scan_kernel<<<B_SZ * NH, 256, 0, stream>>>(zx, dt_b, A_log, Dv, yb);
  gate_kernel<<<ROWS, 256, 0, stream>>>(zx, yb, norm_w, yn);
  gemm_bf16f32<<<dim3(DM / 128, ROWS / 128), 256, 0, stream>>>(yn, W_out, attn, ROWS, DM, DI);
  gemm_bf16f32<<<dim3(INTER / 128, ROWS / 128), 256, 0, stream>>>(hsln, fc1_w, mlph, ROWS, INTER, DM);
  gelu_kernel<<<(ROWS * (size_t)INTER / 4 + 255) / 256, 256, 0, stream>>>(mlph, fc1_b, mlpb);
  gemm_bf16f32<<<dim3(DM / 128, ROWS / 128), 256, 0, stream>>>(mlpb, fc2_w, mlpo, ROWS, DM, INTER);
  final_kernel<<<(ROWS * (size_t)DM / 4 + 255) / 256, 256, 0, stream>>>(attn, mlpo, fc2_b, hs, out);
}
